// localBlock_78374563218047
// MI455X (gfx1250) — compile-verified, matrix operand order checked by reading, not run on hardware
//
#include <hip/hip_runtime.h>

// ---------------- problem constants ----------------
constexpr int Bn = 8;
constexpr int T  = 4096;
constexpr int D  = 512;
constexpr int H  = 8;
constexpr int Wn = 128;          // one-sided window
constexpr int HD = 64;           // head dim
constexpr int C  = T / Wn;       // 32 chunks
constexpr int KW = 3 * Wn;       // 384 keys per window

typedef __bf16 bf16;
typedef bf16  v16bf  __attribute__((ext_vector_type(16)));
typedef bf16  bf16x8 __attribute__((ext_vector_type(8)));
typedef bf16  bf16x4 __attribute__((ext_vector_type(4)));
typedef bf16  bf16x2 __attribute__((ext_vector_type(2)));
typedef float v8f    __attribute__((ext_vector_type(8)));
typedef float f32x4  __attribute__((ext_vector_type(4)));
typedef int   v4i    __attribute__((ext_vector_type(4)));

// ---------------- CDNA5 async Global<->LDS path (guarded) ----------------
#if defined(__HIP_DEVICE_COMPILE__) && defined(__gfx1250__) &&                  \
    __has_builtin(__builtin_amdgcn_global_load_async_to_lds_b128) &&            \
    __has_builtin(__builtin_amdgcn_global_store_async_from_lds_b128) &&         \
    __has_builtin(__builtin_amdgcn_s_wait_asynccnt)
#define USE_ASYNC 1
#else
#define USE_ASYNC 0
#endif

#if USE_ASYNC
// builtin signature: (int4 AS1* global, int4 AS3* lds, imm offset, imm cpol)
typedef v4i __attribute__((address_space(1))) as1_v4i;
typedef v4i __attribute__((address_space(3))) as3_v4i;
__device__ __forceinline__ void async_load16(const bf16* g, bf16* l) {
    __builtin_amdgcn_global_load_async_to_lds_b128((as1_v4i*)g, (as3_v4i*)l, 0, 0);
}
__device__ __forceinline__ void async_store16(bf16* g, const bf16* l) {
    __builtin_amdgcn_global_store_async_from_lds_b128((as1_v4i*)g, (as3_v4i*)l, 0, 0);
}
__device__ __forceinline__ void async_wait0() { __builtin_amdgcn_s_wait_asynccnt(0); }
#else
__device__ __forceinline__ void async_load16(const bf16* g, bf16* l) {
    *(bf16x8*)l = *(const bf16x8*)g;
}
__device__ __forceinline__ void async_store16(bf16* g, const bf16* l) {
    *(bf16x8*)g = *(const bf16x8*)l;
}
__device__ __forceinline__ void async_wait0() {}
#endif

// ---- WMMA helpers ----
__device__ __forceinline__ v8f wmma_bf16(v16bf a, v16bf b, v8f c) {
    return __builtin_amdgcn_wmma_f32_16x16x32_bf16(false, a, false, b, (short)0, c, false, false);
}

// 16-bit operand fragment: lanes 0-15 take K offsets {0..7,16..23},
// lanes 16-31 take {8..15,24..31}. Two contiguous 16B runs -> ds_load_b128 x2.
__device__ __forceinline__ v16bf load_frag(const bf16* rowbase, int half) {
    v16bf f;
    const int o0 = half * 8;
    const int o1 = 16 + half * 8;
#pragma unroll
    for (int j = 0; j < 8; ++j) {
        f[j]     = rowbase[o0 + j];
        f[8 + j] = rowbase[o1 + j];
    }
    return f;
}

// =====================================================================
// Kernel 1: fused QKV projection GEMM (x[32768x512] @ W[512x512] + b)
//   z=0 -> Q * (1/sqrt(HD)) * log2(e)  -> [B,H,T,HD] bf16 (log2-domain scores)
//   z=1 -> K                           -> [B,H,T,HD] bf16
//   z=2 -> V                           -> [B,H,HD,T] bf16 (transposed)
// =====================================================================
__global__ __launch_bounds__(256)
void localBlock_qkv_proj(const float* __restrict__ x,
                         const float* __restrict__ Wq, const float* __restrict__ bq,
                         const float* __restrict__ Wk, const float* __restrict__ bk,
                         const float* __restrict__ Wv, const float* __restrict__ bv,
                         bf16* __restrict__ Qo, bf16* __restrict__ Ko, bf16* __restrict__ Vo) {
    const int which = blockIdx.z;
    const float* Wm = (which == 0) ? Wq : (which == 1) ? Wk : Wv;
    const float* bm = (which == 0) ? bq : (which == 1) ? bk : bv;

    const int m0 = blockIdx.x * 128;
    const int n0 = blockIdx.y * 64;

    // staging tiles and the C bounce tile share the same LDS (both 18432 B)
    __shared__ union SM {
        struct { bf16 A[128][48]; bf16 Bt[64][48]; } s;   // 32-wide K slices
        bf16 Cs[128][72];                                 // 128 x 64 results
    } sm;

    const int tid  = threadIdx.x;
    const int lane = tid & 31;
    const int l16  = lane & 15;
    const int half = lane >> 4;
    const int w    = tid >> 5;

    v8f acc0 = {}, acc1 = {}, acc2 = {}, acc3 = {};

    for (int k0 = 0; k0 < D; k0 += 32) {
        // stage A: float4 global loads -> packed bf16x4 DS stores (4 per thread)
#pragma unroll
        for (int u = tid; u < 128 * 8; u += 256) {
            int r = u >> 3, c4 = (u & 7) * 4;
            f32x4 v = *(const f32x4*)&x[(size_t)(m0 + r) * D + k0 + c4];
            bf16x4 p;
#pragma unroll
            for (int i = 0; i < 4; ++i) p[i] = (bf16)v[i];
            *(bf16x4*)&sm.s.A[r][c4] = p;
        }
        // stage B^T: n-fastest (coalesced) row-pair loads -> packed bf16x2 stores
#pragma unroll
        for (int u = tid; u < 64 * 16; u += 256) {
            int n = u & 63, kp = u >> 6;                  // kp = 0..15 -> kk = 2*kp
            float a0 = Wm[(size_t)(k0 + 2 * kp) * D + n0 + n];
            float a1 = Wm[(size_t)(k0 + 2 * kp + 1) * D + n0 + n];
            bf16x2 p;
            p[0] = (bf16)a0;
            p[1] = (bf16)a1;
            *(bf16x2*)&sm.s.Bt[n][2 * kp] = p;
        }
        __syncthreads();

        v16bf af  = load_frag(&sm.s.A[w * 16 + l16][0], half);
        v16bf bf0 = load_frag(&sm.s.Bt[0  + l16][0], half);
        acc0 = wmma_bf16(af, bf0, acc0);
        v16bf bf1 = load_frag(&sm.s.Bt[16 + l16][0], half);
        acc1 = wmma_bf16(af, bf1, acc1);
        v16bf bf2 = load_frag(&sm.s.Bt[32 + l16][0], half);
        acc2 = wmma_bf16(af, bf2, acc2);
        v16bf bf3 = load_frag(&sm.s.Bt[48 + l16][0], half);
        acc3 = wmma_bf16(af, bf3, acc3);

        __syncthreads();
    }

    // ---- epilogue: bias (+ scale for Q), bounce through LDS, coalesced out
    const float qscale = 0.125f * 1.4426950408889634f;   // 1/sqrt(64) * log2(e)
#pragma unroll
    for (int nt = 0; nt < 4; ++nt) {
        const v8f accv = (nt == 0) ? acc0 : (nt == 1) ? acc1 : (nt == 2) ? acc2 : acc3;
        const float bv = bm[n0 + nt * 16 + l16];
#pragma unroll
        for (int r = 0; r < 8; ++r) {
            float val = accv[r] + bv;
            if (which == 0) val *= qscale;
            sm.Cs[w * 16 + r + half * 8][nt * 16 + l16] = (bf16)val;
        }
    }
    __syncthreads();

    const int hh = n0 >> 6;                       // head (tile never crosses heads)
    if (which < 2) {
        bf16* outp = (which == 0) ? Qo : Ko;
        // 128 rows x 8 b128 units, LDS -> global async DMA
#pragma unroll 2
        for (int u = tid; u < 128 * 8; u += 256) {
            int r = u >> 3, c8 = (u & 7) * 8;
            int m = m0 + r;
            int bb = m >> 12, t = m & (T - 1);
            async_store16(outp + (((size_t)bb * H + hh) * T + t) * HD + c8, &sm.Cs[r][c8]);
        }
    } else {
        // V^T: gather LDS columns, 8B vector stores contiguous along t
        const int bb = m0 >> 12, tbase = m0 & (T - 1);
#pragma unroll 2
        for (int u = tid; u < 64 * 32; u += 256) {
            int hd = u >> 5, q4 = (u & 31) * 4;
            bf16x4 tmp;
#pragma unroll
            for (int i = 0; i < 4; ++i) tmp[i] = sm.Cs[q4 + i][hd];
            *(bf16x4*)(Vo + (((size_t)bb * H + hh) * HD + hd) * T + tbase + q4) = tmp;
        }
    }
    async_wait0();
}

// =====================================================================
// Kernel 2: banded local attention per (b,h,chunk) + residual.
// 8 wave32; flash-style online softmax (log2 domain) over 12 key blocks,
// double-buffered K/V staged by async Global->LDS DMA.
// =====================================================================
__device__ __forceinline__ void stage_kv(bf16 (*Kbuf)[80], bf16 (*Vbuf)[48],
                                         const bf16* __restrict__ Kg,
                                         const bf16* __restrict__ Vg,
                                         int t0, int kb, int tid) {
    const int tblk = t0 + kb;                   // 32-key blocks are uniformly in/out of range
    const bool ok = (tblk >= 0) && (tblk < T);
    {   // K block: 32 keys x 64 hd; one b128 per thread
        const int kk = tid >> 3, c8 = (tid & 7) * 8;
        bf16* l = &Kbuf[kk][c8];
        if (ok) {
            async_load16(Kg + (size_t)(tblk + kk) * HD + c8, l);
        } else {
            bf16x8 z = {};
            *(bf16x8*)l = z;
        }
    }
    {   // V^T block: 64 hd x 32 keys; one b128 per thread
        const int dd = tid >> 2, c8 = (tid & 3) * 8;
        bf16* l = &Vbuf[dd][c8];
        if (ok) {
            async_load16(Vg + (size_t)dd * T + tblk + c8, l);
        } else {
            bf16x8 z = {};
            *(bf16x8*)l = z;
        }
    }
}

__global__ __launch_bounds__(256)
void localBlock_attn(const bf16* __restrict__ Q, const bf16* __restrict__ K,
                     const bf16* __restrict__ VT, const float* __restrict__ x,
                     float* __restrict__ out) {
    const int cid = blockIdx.x;
    const int c = cid & (C - 1);
    const int h = (cid / C) & (H - 1);
    const int b = cid / (C * H);

    const int tid  = threadIdx.x;
    const int lane = tid & 31;
    const int l16  = lane & 15;
    const int half = lane >> 4;
    const int w    = tid >> 5;

    __shared__ bf16 Qs[128][80];      // 20.0 KB
    __shared__ bf16 Ks[2][32][80];    // 10.0 KB (double-buffered)
    __shared__ bf16 Vs[2][64][48];    // 12.0 KB (double-buffered)
    __shared__ bf16 Ps[8][16][48];    // 12.0 KB per-wave P bounce

    const size_t bh = (size_t)b * H + h;
    const bf16* Qg = Q  + (bh * T + (size_t)c * Wn) * HD;
    const bf16* Kg = K  + bh * (size_t)T * HD;
    const bf16* Vg = VT + bh * (size_t)HD * T;
    const int t0 = c * Wn - Wn;       // global t at window position j=0

    // stage Q chunk (async DMA, 4 b128 units per thread) + first K/V block
#pragma unroll 2
    for (int u = tid; u < 128 * 8; u += 256) {
        int r = u >> 3, c8 = (u & 7) * 8;
        async_load16(Qg + (size_t)r * HD + c8, &Qs[r][c8]);
    }
    stage_kv(Ks[0], Vs[0], Kg, Vg, t0, 0, tid);
    async_wait0();
    __syncthreads();

    const v16bf qf0 = load_frag(&Qs[w * 16 + l16][0],  half);
    const v16bf qf1 = load_frag(&Qs[w * 16 + l16][32], half);

    v8f o0 = {}, o1 = {}, o2 = {}, o3 = {};
    float mrow[8], lrow[8];
#pragma unroll
    for (int r = 0; r < 8; ++r) { mrow[r] = -3.0e38f; lrow[r] = 0.0f; }

    int ib = 0;
    for (int kb = 0; kb < KW; kb += 32, ib ^= 1) {
        // issue async copy of next block into the other buffer
        if (kb + 32 < KW) stage_kv(Ks[ib ^ 1], Vs[ib ^ 1], Kg, Vg, t0, kb + 32, tid);

        const int plo = w * 16;
        const bool active = (kb + 31 >= plo) && (kb <= plo + 15 + 2 * Wn);
        if (active) {
            // ---- S[16x32] = Qw(16x64) . K^T(64x32): 4 WMMAs
            v8f s0 = {}, s1 = {};
            {
                v16bf b00 = load_frag(&Ks[ib][0  + l16][0],  half);
                s0 = wmma_bf16(qf0, b00, s0);
                v16bf b01 = load_frag(&Ks[ib][0  + l16][32], half);
                s0 = wmma_bf16(qf1, b01, s0);
                v16bf b10 = load_frag(&Ks[ib][16 + l16][0],  half);
                s1 = wmma_bf16(qf0, b10, s1);
                v16bf b11 = load_frag(&Ks[ib][16 + l16][32], half);
                s1 = wmma_bf16(qf1, b11, s1);
            }

            // ---- band mask + online softmax in exp2 domain
            const int j0 = kb + l16;
            const int j1 = kb + 16 + l16;
            const int g0 = t0 + j0;
            const int g1 = t0 + j1;
#pragma unroll
            for (int r = 0; r < 8; ++r) {
                const int p = w * 16 + r + half * 8;
                const bool v0 = (j0 >= p) && (j0 <= p + 2 * Wn) && (g0 >= 0) && (g0 < T);
                const bool v1 = (j1 >= p) && (j1 <= p + 2 * Wn) && (g1 >= 0) && (g1 < T);
                const float e0 = v0 ? s0[r] : -3.0e38f;
                const float e1 = v1 ? s1[r] : -3.0e38f;

                float bmx = fmaxf(e0, e1);
                bmx = fmaxf(bmx, __shfl_xor(bmx, 8, 32));
                bmx = fmaxf(bmx, __shfl_xor(bmx, 4, 32));
                bmx = fmaxf(bmx, __shfl_xor(bmx, 2, 32));
                bmx = fmaxf(bmx, __shfl_xor(bmx, 1, 32));

                const float mnew = fmaxf(mrow[r], bmx);
                const float sc   = __builtin_amdgcn_exp2f(mrow[r] - mnew);
                const float p0   = v0 ? __builtin_amdgcn_exp2f(s0[r] - mnew) : 0.0f;
                const float p1   = v1 ? __builtin_amdgcn_exp2f(s1[r] - mnew) : 0.0f;

                float rs = p0 + p1;
                rs += __shfl_xor(rs, 8, 32);
                rs += __shfl_xor(rs, 4, 32);
                rs += __shfl_xor(rs, 2, 32);
                rs += __shfl_xor(rs, 1, 32);

                mrow[r] = mnew;
                lrow[r] = lrow[r] * sc + rs;
                o0[r] *= sc; o1[r] *= sc; o2[r] *= sc; o3[r] *= sc;

                Ps[w][r + half * 8][l16]      = (bf16)p0;
                Ps[w][r + half * 8][16 + l16] = (bf16)p1;
            }

            // ---- O += P(16x32) . V(32x64): 4 WMMAs (same-wave DS in order)
            const v16bf pf = load_frag(&Ps[w][l16][0], half);
            v16bf vf0 = load_frag(&Vs[ib][0  + l16][0], half);
            o0 = wmma_bf16(pf, vf0, o0);
            v16bf vf1 = load_frag(&Vs[ib][16 + l16][0], half);
            o1 = wmma_bf16(pf, vf1, o1);
            v16bf vf2 = load_frag(&Vs[ib][32 + l16][0], half);
            o2 = wmma_bf16(pf, vf2, o2);
            v16bf vf3 = load_frag(&Vs[ib][48 + l16][0], half);
            o3 = wmma_bf16(pf, vf3, o3);
        }

        async_wait0();       // next buffer resident before anyone passes the barrier
        __syncthreads();
    }

    // ---- epilogue: normalize, residual, fp32 out
#pragma unroll
    for (int r = 0; r < 8; ++r) {
        const float inv = 1.0f / lrow[r];
        const int qlocal = w * 16 + r + half * 8;
        const int t = c * Wn + qlocal;
        const size_t base = ((size_t)b * T + t) * D + (size_t)h * HD;
        out[base + 0  + l16] = o0[r] * inv + x[base + 0  + l16];
        out[base + 16 + l16] = o1[r] * inv + x[base + 16 + l16];
        out[base + 32 + l16] = o2[r] * inv + x[base + 32 + l16];
        out[base + 48 + l16] = o3[r] * inv + x[base + 48 + l16];
    }
}

// =====================================================================
extern "C" void kernel_launch(void* const* d_in, const int* in_sizes, int n_in,
                              void* d_out, int out_size, void* d_ws, size_t ws_size,
                              hipStream_t stream) {
    const float* x  = (const float*)d_in[0];
    const float* Wq = (const float*)d_in[1];
    const float* bq = (const float*)d_in[2];
    const float* Wk = (const float*)d_in[3];
    const float* bk = (const float*)d_in[4];
    const float* Wv = (const float*)d_in[5];
    const float* bv = (const float*)d_in[6];
    float* out = (float*)d_out;

    const size_t elems = (size_t)Bn * H * T * HD;   // 16.7M per tensor
    bf16* Qw = (bf16*)d_ws;
    bf16* Kw = Qw + elems;
    bf16* Vw = Kw + elems;

    dim3 gProj((Bn * T) / 128, D / 64, 3);          // 256 x 8 x 3 blocks
    localBlock_qkv_proj<<<gProj, 256, 0, stream>>>(x, Wq, bq, Wk, bk, Wv, bv, Qw, Kw, Vw);

    dim3 gAttn(Bn * H * C);                         // 2048 blocks
    localBlock_attn<<<gAttn, 256, 0, stream>>>(Qw, Kw, Vw, x, out);
}
